// KernelNN_42898133352779
// MI455X (gfx1250) — matrix operand order repaired and checked by reading, not compiled, not run on hardware
//
#include <hip/hip_runtime.h>

typedef __attribute__((ext_vector_type(16))) __bf16 v16bf;
typedef __attribute__((ext_vector_type(8)))  float  v8f;

// ---------------- prep kernels ----------------
__global__ void cvt_f32_to_bf16(const float* __restrict__ in, __bf16* __restrict__ out, int n){
  int i = blockIdx.x * blockDim.x + threadIdx.x;
  if (i < n) out[i] = (__bf16)in[i];
}

__global__ void transpose_root_bf16(const float* __restrict__ root, __bf16* __restrict__ rootT){
  int i = threadIdx.x;                 // one block of 1024
  int r = i >> 5, c = i & 31;
  rootT[c * 32 + r] = (__bf16)root[r * 32 + c];   // rootT[o][i] = root[i][o]
}

// h1_pre[e,c] = edge_attr[e,:6] . k1_w[c,:6] + k1_b[c]
__global__ void k1_kernel(const float* __restrict__ ea, const float* __restrict__ w,
                          const float* __restrict__ b, float* __restrict__ out, int E){
  __shared__ float e6[8];
  int e = blockIdx.x;
  if (threadIdx.x < 6) e6[threadIdx.x] = ea[(size_t)e * 6 + threadIdx.x];
  __syncthreads();
  int c = threadIdx.x;                 // 256 threads = 256 output cols
  float acc = b[c];
  #pragma unroll
  for (int i = 0; i < 6; ++i) acc = fmaf(e6[i], w[c * 6 + i], acc);
  out[(size_t)e * 256 + c] = acc;
}

// column sums / sums of squares over rows (cols fixed = 256)
__global__ void colstats_kernel(const float* __restrict__ X, int rows,
                                float* __restrict__ sum, float* __restrict__ sumsq){
  const int c = threadIdx.x;           // 256 threads = 256 cols
  int r0 = blockIdx.x * 512;
  int r1 = min(r0 + 512, rows);
  float s = 0.f, q = 0.f;
  for (int r = r0; r < r1; ++r){
    float v = X[(size_t)r * 256 + c];
    s += v; q = fmaf(v, v, q);
  }
  atomicAdd(&sum[c], s);
  atomicAdd(&sumsq[c], q);
}

__global__ void bn_finalize_kernel(const float* __restrict__ sum, const float* __restrict__ sq,
                                   const float* __restrict__ g, const float* __restrict__ b,
                                   float* __restrict__ scale, float* __restrict__ shift, int rows){
  int c = threadIdx.x;
  float inv = 1.0f / (float)rows;
  float mu  = sum[c] * inv;
  float var = sq[c] * inv - mu * mu;
  float s   = g[c] * rsqrtf(var + 1e-5f);
  scale[c] = s;
  shift[c] = b[c] - mu * s;
}

// ---------------- WMMA GEMM: out[e,o] = relu(BN(Apre))[e,:] . Wt[o,:] + bias[o] ----------------
// A tile (16 edges x K=256) in registers; B tiles software-pipelined through LDS (regs for t+1
// fetched while tile t computes). Requires E % 64 == 0, Cout % 16 == 0, Cin == 256.
template<bool BF16_OUT>
__global__ void __launch_bounds__(128)
gemm_bn_relu_wmma(const float* __restrict__ Apre,
                  const float* __restrict__ scale,
                  const float* __restrict__ shift,
                  const __bf16* __restrict__ Wt,      // [Cout, 256] bf16 row-major
                  const float* __restrict__ bias,
                  void* __restrict__ outp,
                  int Cout)
{
  constexpr int Cin = 256;
  __shared__ __align__(32) __bf16 Bt[16 * Cin];       // 16-col x 256-K B tile (8 KB)
  __shared__ float s_scale[Cin];
  __shared__ float s_shift[Cin];
  const int tid  = threadIdx.x;
  const int lane = tid & 31;
  const int wib  = tid >> 5;                          // wave in block (0..3)
  const int eb   = blockIdx.x * 64 + wib * 16;        // edge-tile base
  const bool hi  = lane >= 16;
  const float* ar = Apre + (size_t)(eb + (lane & 15)) * Cin;

  s_scale[tid]       = scale[tid];
  s_scale[tid + 128] = scale[tid + 128];
  s_shift[tid]       = shift[tid];
  s_shift[tid + 128] = shift[tid + 128];
  __syncthreads();

  // Load full A tile once: BN scale/shift + ReLU + cvt to bf16, contiguous k = 16*h + s per lane half.
  v16bf a[8];
  #pragma unroll
  for (int kc = 0; kc < 8; ++kc){
    const int ka = kc * 32 + (hi ? 16 : 0);
    #pragma unroll
    for (int t = 0; t < 8; ++t){
      float v0 = fmaxf(fmaf(ar[ka + t],      s_scale[ka + t],      s_shift[ka + t]),      0.f);
      float v1 = fmaxf(fmaf(ar[ka + 8 + t], s_scale[ka + 8 + t], s_shift[ka + 8 + t]), 0.f);
      a[kc][t]     = (__bf16)v0;
      a[kc][t + 8] = (__bf16)v1;
    }
  }

  const int jcol = lane & 31 & 15;
  const int nt = Cout >> 4;

  // prefetch B tile 0 into registers
  int4 breg[4];
  {
    const int4* s4 = (const int4*)Wt;
    #pragma unroll
    for (int q = 0; q < 4; ++q) breg[q] = s4[q * 128 + tid];
  }

  for (int to = 0; to < nt; ++to){
    __syncthreads();                                  // LDS from previous tile consumed
    {
      int4* d4 = (int4*)Bt;
      #pragma unroll
      for (int q = 0; q < 4; ++q) d4[q * 128 + tid] = breg[q];
    }
    __syncthreads();
    if (to + 1 < nt){                                 // overlap next fetch with compute
      const int4* s4 = (const int4*)(Wt + (size_t)(to + 1) * 16 * Cin);
      #pragma unroll
      for (int q = 0; q < 4; ++q) breg[q] = s4[q * 128 + tid];
    }

    v8f acc = {};
    #pragma unroll
    for (int kc = 0; kc < 8; ++kc){
      const int kb = kc * 32 + (hi ? 16 : 0);
      v16bf bfrag = *(const v16bf*)(&Bt[jcol * Cin + kb]);
      acc = __builtin_amdgcn_wmma_f32_16x16x32_bf16(false, a[kc], false, bfrag,
                                                    (short)0, acc, false, false);
    }

    const int ncol = to * 16 + jcol;
    const float bi = bias[ncol];
    if (BF16_OUT){
      __bf16* ob = (__bf16*)outp;
      #pragma unroll
      for (int r = 0; r < 8; ++r){
        const int m = eb + r + (hi ? 8 : 0);
        ob[(size_t)m * Cout + ncol] = (__bf16)(acc[r] + bi);
      }
    } else {
      float* of = (float*)outp;
      #pragma unroll
      for (int r = 0; r < 8; ++r){
        const int m = eb + r + (hi ? 8 : 0);
        of[(size_t)m * Cout + ncol] = acc[r] + bi;
      }
    }
  }
}

// ---------------- node pipeline ----------------
__global__ void fc1_kernel(const float* __restrict__ x, const float* __restrict__ w,
                           const float* __restrict__ b, float* __restrict__ h, int N){
  int id = blockIdx.x * blockDim.x + threadIdx.x;
  if (id >= N * 32) return;
  int n = id >> 5, c = id & 31;
  float acc = b[c];
  #pragma unroll
  for (int i = 0; i < 6; ++i) acc = fmaf(x[(size_t)n * 6 + i], w[c * 6 + i], acc);
  h[id] = acc;
}

__global__ void deg_kernel(const int* __restrict__ dst, float* __restrict__ deg, int E){
  int e = blockIdx.x * blockDim.x + threadIdx.x;
  if (e < E) atomicAdd(&deg[dst[e]], 1.0f);
}

__global__ void invdeg_kernel(float* __restrict__ deg, int N){
  int i = blockIdx.x * blockDim.x + threadIdx.x;
  if (i < N) deg[i] = 1.0f / fmaxf(deg[i], 1.0f);
}

// msg[e,o] = sum_i h[src[e],i] * W_edge[e,i,o]; atomically aggregate into agg[dst[e],o]
// One wave per edge; lane = output o; streams 2 KB of bf16 W per edge, coalesced 64B per i.
__global__ void msg_kernel(const int* __restrict__ ei, const float* __restrict__ hnode,
                           const __bf16* __restrict__ W, float* __restrict__ agg, int E){
  const int wid  = (int)((blockIdx.x * (size_t)blockDim.x + threadIdx.x) >> 5);
  const int lane = threadIdx.x & 31;
  if (wid >= E) return;
  const int s = ei[wid];
  const int d = ei[E + wid];
  const float h = hnode[(size_t)s * 32 + lane];
  const __bf16* We = W + (size_t)wid * 1024;
  __builtin_prefetch((const void*)(We + 512), 0, 1);  // warm tail of this edge's 2KB row
  float m = 0.f;
  #pragma unroll 8
  for (int i = 0; i < 32; ++i){
    float hb = __shfl(h, i, 32);
    m = fmaf(hb, (float)We[i * 32 + lane], m);
  }
  atomicAdd(&agg[(size_t)d * 32 + lane], m);
}

// hnext = relu(agg*invdeg + hnode @ root + conv_bias), hnode@root via bf16 WMMA
__global__ void update_wmma_kernel(const float* __restrict__ hnode,
                                   const __bf16* __restrict__ rootT,   // [32(o)][32(i)]
                                   const float* __restrict__ agg,
                                   const float* __restrict__ invdeg,
                                   const float* __restrict__ cbias,
                                   float* __restrict__ hnext, int Ntiles){
  const int wave = (int)((blockIdx.x * (size_t)blockDim.x + threadIdx.x) >> 5);
  const int lane = threadIdx.x & 31;
  const int tn = wave >> 1;
  const int to = wave & 1;
  if (tn >= Ntiles) return;
  const bool hi = lane >= 16;
  const int row = tn * 16 + (lane & 15);
  const float* ar = hnode + (size_t)row * 32;
  const int ka = hi ? 16 : 0;
  v16bf a;
  #pragma unroll
  for (int t = 0; t < 8; ++t){
    a[t]     = (__bf16)ar[ka + t];
    a[t + 8] = (__bf16)ar[ka + 8 + t];
  }
  const int ocol = to * 16 + (lane & 15);
  v16bf b = *(const v16bf*)(rootT + ocol * 32 + (hi ? 16 : 0));
  v8f acc = {};
  acc = __builtin_amdgcn_wmma_f32_16x16x32_bf16(false, a, false, b, (short)0, acc, false, false);
  const float cb = cbias[ocol];
  #pragma unroll
  for (int r = 0; r < 8; ++r){
    const int m = tn * 16 + r + (hi ? 8 : 0);
    const float v = acc[r] + agg[(size_t)m * 32 + ocol] * invdeg[m] + cb;
    hnext[(size_t)m * 32 + ocol] = fmaxf(v, 0.f);
  }
}

__global__ void fc2_kernel(const float* __restrict__ h, const float* __restrict__ w,
                           const float* __restrict__ b, float* __restrict__ out, int N){
  const int wid  = (int)((blockIdx.x * (size_t)blockDim.x + threadIdx.x) >> 5);
  const int lane = threadIdx.x & 31;
  if (wid >= N) return;
  float p = h[(size_t)wid * 32 + lane] * w[lane];
  #pragma unroll
  for (int off = 16; off > 0; off >>= 1) p += __shfl_xor(p, off, 32);
  if (lane == 0) out[wid] = p + b[0];
}

// ---------------- host ----------------
extern "C" void kernel_launch(void* const* d_in, const int* in_sizes, int n_in,
                              void* d_out, int out_size, void* d_ws, size_t ws_size,
                              hipStream_t stream) {
  const float* x     = (const float*)d_in[0];
  const int*   eidx  = (const int*)  d_in[1];
  const float* eattr = (const float*)d_in[2];
  const float* fc1w  = (const float*)d_in[3];
  const float* fc1b  = (const float*)d_in[4];
  const float* k1w   = (const float*)d_in[5];
  const float* k1b   = (const float*)d_in[6];
  const float* bn1g  = (const float*)d_in[7];
  const float* bn1b  = (const float*)d_in[8];
  const float* k2w   = (const float*)d_in[9];
  const float* k2b   = (const float*)d_in[10];
  const float* bn2g  = (const float*)d_in[11];
  const float* bn2b  = (const float*)d_in[12];
  const float* k3w   = (const float*)d_in[13];
  const float* k3b   = (const float*)d_in[14];
  const float* root  = (const float*)d_in[15];
  const float* cbias = (const float*)d_in[16];
  const float* fc2w  = (const float*)d_in[17];
  const float* fc2b  = (const float*)d_in[18];
  float* out = (float*)d_out;
  (void)n_in; (void)out_size; (void)ws_size;

  const int N = in_sizes[0] / 6;      // 20000
  const int E = in_sizes[1] / 2;      // 200000

  char* ws = (char*)d_ws;
  size_t off = 0;
  auto alloc = [&](size_t bytes) -> void* {
    void* p = ws + off;
    off = (off + bytes + 255) & ~(size_t)255;
    return p;
  };
  float*  h1     = (float*) alloc((size_t)E * 256 * sizeof(float));
  float*  h2     = (float*) alloc((size_t)E * 256 * sizeof(float));
  __bf16* Wedge  = (__bf16*)alloc((size_t)E * 1024 * sizeof(__bf16));
  float*  hA     = (float*) alloc((size_t)N * 32 * sizeof(float));
  float*  hB     = (float*) alloc((size_t)N * 32 * sizeof(float));
  float*  agg    = (float*) alloc((size_t)N * 32 * sizeof(float));
  float*  deg    = (float*) alloc((size_t)N * sizeof(float));
  __bf16* k2bf   = (__bf16*)alloc(256 * 256 * sizeof(__bf16));
  __bf16* k3bf   = (__bf16*)alloc(1024 * 256 * sizeof(__bf16));
  __bf16* rootT  = (__bf16*)alloc(32 * 32 * sizeof(__bf16));
  float*  sum1   = (float*) alloc(256 * sizeof(float));
  float*  sq1    = (float*) alloc(256 * sizeof(float));
  float*  scale1 = (float*) alloc(256 * sizeof(float));
  float*  shift1 = (float*) alloc(256 * sizeof(float));
  float*  sum2   = (float*) alloc(256 * sizeof(float));
  float*  sq2    = (float*) alloc(256 * sizeof(float));
  float*  scale2 = (float*) alloc(256 * sizeof(float));
  float*  shift2 = (float*) alloc(256 * sizeof(float));

  // weight prep (bf16 copies; root transposed so B columns are contiguous)
  cvt_f32_to_bf16<<<(256 * 256 + 255) / 256, 256, 0, stream>>>(k2w, k2bf, 256 * 256);
  cvt_f32_to_bf16<<<(1024 * 256 + 255) / 256, 256, 0, stream>>>(k3w, k3bf, 1024 * 256);
  transpose_root_bf16<<<1, 1024, 0, stream>>>(root, rootT);

  // edge MLP stage 1 + BN1 stats
  k1_kernel<<<E, 256, 0, stream>>>(eattr, k1w, k1b, h1, E);
  hipMemsetAsync(sum1, 0, 256 * sizeof(float), stream);
  hipMemsetAsync(sq1,  0, 256 * sizeof(float), stream);
  colstats_kernel<<<(E + 511) / 512, 256, 0, stream>>>(h1, E, sum1, sq1);
  bn_finalize_kernel<<<1, 256, 0, stream>>>(sum1, sq1, bn1g, bn1b, scale1, shift1, E);

  // stage 2: h2 = relu(BN1(h1)) @ k2^T + k2_b   (bf16 WMMA, f32 out)
  gemm_bn_relu_wmma<false><<<E / 64, 128, 0, stream>>>(h1, scale1, shift1, k2bf, k2b,
                                                       (void*)h2, 256);

  hipMemsetAsync(sum2, 0, 256 * sizeof(float), stream);
  hipMemsetAsync(sq2,  0, 256 * sizeof(float), stream);
  colstats_kernel<<<(E + 511) / 512, 256, 0, stream>>>(h2, E, sum2, sq2);
  bn_finalize_kernel<<<1, 256, 0, stream>>>(sum2, sq2, bn2g, bn2b, scale2, shift2, E);

  // stage 3: W_edge = relu(BN2(h2)) @ k3^T + k3_b   (bf16 WMMA, bf16 out halves stream BW)
  gemm_bn_relu_wmma<true><<<E / 64, 128, 0, stream>>>(h2, scale2, shift2, k3bf, k3b,
                                                      (void*)Wedge, 1024);

  // node features + degree
  fc1_kernel<<<(N * 32 + 255) / 256, 256, 0, stream>>>(x, fc1w, fc1b, hA, N);
  hipMemsetAsync(deg, 0, (size_t)N * sizeof(float), stream);
  deg_kernel<<<(E + 255) / 256, 256, 0, stream>>>(eidx + E, deg, E);
  invdeg_kernel<<<(N + 255) / 256, 256, 0, stream>>>(deg, N);

  // depth loop
  float* hcur = hA; float* hnxt = hB;
  const int Ntiles = N / 16;
  for (int d = 0; d < 3; ++d){
    hipMemsetAsync(agg, 0, (size_t)N * 32 * sizeof(float), stream);
    msg_kernel<<<(int)(((size_t)E * 32 + 255) / 256), 256, 0, stream>>>(eidx, hcur, Wedge, agg, E);
    update_wmma_kernel<<<(Ntiles * 2 * 32 + 255) / 256, 256, 0, stream>>>(
        hcur, rootT, agg, deg, cbias, hnxt, Ntiles);
    float* t = hcur; hcur = hnxt; hnxt = t;
  }

  fc2_kernel<<<(int)(((size_t)N * 32 + 255) / 256), 256, 0, stream>>>(hcur, fc2w, fc2b, out, N);
}
